// RNNbase_11562051961101
// MI455X (gfx1250) — matrix operand order repaired and checked by reading, not compiled, not run on hardware
//
#include <hip/hip_runtime.h>
#include <hip/hip_bf16.h>

// ---------------- problem constants ----------------
#define SEQ    128
#define BATCH  64
#define VOCAB  10000
#define HID    1024
#define ROWS   (SEQ * BATCH)        // 8192
#define STEPEL (BATCH * HID)        // 65536 elements per timestep slice

typedef __attribute__((ext_vector_type(16))) __bf16 v16bf;
typedef __attribute__((ext_vector_type(8)))  __bf16 v8bf;
typedef __attribute__((ext_vector_type(8)))  float  v8f;

// ---------------- WMMA helpers ----------------
// CDNA5 bf16 WMMA: D(16x16 f32) = A(16x32 bf16) * B(32x16 bf16) + C
__device__ __forceinline__ v8f wmma_bf16(v16bf a, v16bf b, v8f c) {
  return __builtin_amdgcn_wmma_f32_16x16x32_bf16(
      /*neg_a=*/false, a, /*neg_b=*/false, b,
      /*c_mod=*/(short)0, c, /*reuse_a=*/false, /*reuse_b=*/false);
}

// Load 16 bf16 elements as two aligned 16-byte chunks.
__device__ __forceinline__ v16bf load_frag16(const __bf16* p0, const __bf16* p1) {
  v8bf a = *reinterpret_cast<const v8bf*>(p0);
  v8bf b = *reinterpret_cast<const v8bf*>(p1);
  v16bf v;
#pragma unroll
  for (int i = 0; i < 8; ++i) { v[i] = a[i]; v[i + 8] = b[i]; }
  return v;
}

__device__ __forceinline__ v8f zero_acc() {
  v8f v;
#pragma unroll
  for (int i = 0; i < 8; ++i) v[i] = 0.0f;
  return v;
}

// A fragment (16x32, bf16): lanes 0-15 rows M=0..15 with K = k0..k0+15,
// lanes 16-31 same rows with K = k0+16..k0+31.
__device__ __forceinline__ v16bf load_A_frag(const __bf16* A, int row, int lda,
                                             int k0, int half) {
  const __bf16* p = A + (size_t)row * lda + k0 + half * 16;
  return load_frag16(p, p + 8);
}

// B fragment (32x16, bf16) from Bt[N][K]: lane = column n, contiguous K run of 16.
__device__ __forceinline__ v16bf load_B_frag(const __bf16* Bt, int n, int ldb,
                                             int k0, int half) {
  const __bf16* p = Bt + (size_t)n * ldb + k0 + half * 16;
  return load_frag16(p, p + 8);
}

// ---------------- big GEMM: C[M,N] = A[M,K] @ Bt[N,K]^T + bias ----------------
// block = 256 threads (8 waves); block tile 128x128; wave tile 32x64 (2x4 wmma tiles).
// M (gridDim.y*128) divides exactly; N edge handled by clamped loads + guarded stores
// => branch-free inner loop (no exec-mask juggling around the WMMAs).
__global__ void __launch_bounds__(256)
gemm_bf16_f32(const __bf16* __restrict__ A, const __bf16* __restrict__ Bt,
              const float* __restrict__ bias, float* __restrict__ C,
              int N, int K, int ldc) {
  const int tid  = threadIdx.x;
  const int wave = tid >> 5;
  const int lane = tid & 31;
  const int lm   = lane & 15;
  const int half = lane >> 4;
  const int wm   = wave >> 1;                 // 0..3
  const int wn   = wave & 1;                  // 0..1
  const int m_base = blockIdx.y * 128 + wm * 32;
  const int n_base = blockIdx.x * 128 + wn * 64;

  // loop-invariant clamped B columns (always-in-bounds loads)
  int ncol[4];
#pragma unroll
  for (int j = 0; j < 4; ++j) {
    const int n = n_base + j * 16 + lm;
    ncol[j] = (n < N) ? n : (N - 1);
  }

  v8f acc[2][4];
#pragma unroll
  for (int i = 0; i < 2; ++i)
#pragma unroll
    for (int j = 0; j < 4; ++j) acc[i][j] = zero_acc();

  for (int k0 = 0; k0 < K; k0 += 32) {
    v16bf af[2];
#pragma unroll
    for (int i = 0; i < 2; ++i)
      af[i] = load_A_frag(A, m_base + i * 16 + lm, K, k0, half);

    // speculative prefetch of next A K-tile (global_prefetch_b8); OOB is dropped
    __builtin_prefetch(A + (size_t)(m_base + lm) * K + k0 + 32, 0, 3);

    v16bf bfr[4];
#pragma unroll
    for (int j = 0; j < 4; ++j)
      bfr[j] = load_B_frag(Bt, ncol[j], K, k0, half);

#pragma unroll
    for (int i = 0; i < 2; ++i)
#pragma unroll
      for (int j = 0; j < 4; ++j)
        acc[i][j] = wmma_bf16(af[i], bfr[j], acc[i][j]);
  }

#pragma unroll
  for (int i = 0; i < 2; ++i)
#pragma unroll
    for (int j = 0; j < 4; ++j) {
      const int n = n_base + j * 16 + lm;
      if (n >= N) continue;
      const float bia = bias ? bias[n] : 0.0f;
#pragma unroll
      for (int r = 0; r < 8; ++r) {
        const int row = m_base + i * 16 + half * 8 + r;
        C[(size_t)row * ldc + n] = acc[i][j][r] + bia;
      }
    }
}

// ---------------- persistent recurrence: H[t+1] = tanh(Apre[t] + H[t] @ Wh^T) --------
// grid = 32 WGs x 128 threads (4 waves). Each WG owns a 64x32 output tile (all rows,
// 32 columns). Wh tile (32 cols x 1024 K) is staged ONCE into LDS in fragment-packed
// order: ldsB[ktile][jtile][lane][16 bf16] -> each lane's B-fragment is one contiguous
// 32B LDS run; consecutive lanes stride 32B => dense conflict-free ds_load_b128.
// Cross-WG step dependency handled with a device-scope atomic barrier.
__global__ void __launch_bounds__(128)
rnn_recur(const __bf16* __restrict__ WhT, const float* __restrict__ Apre,
          __bf16* __restrict__ H, int* __restrict__ ctr, int nwg) {
  __shared__ __bf16 ldsB[32 * 2 * 32 * 16];   // 64 KB exactly

  const int tid  = threadIdx.x;
  const int wave = tid >> 5;
  const int lane = tid & 31;
  const int lm   = lane & 15;
  const int half = lane >> 4;
  const int m0   = wave * 16;              // 4 waves cover rows 0..63
  const int n0   = blockIdx.x * 32;        // 32 WGs cover cols 0..1023

  // ---- stage Wh tile into LDS (fragment-packed), once ----
  // slot s = ktile*64 + j*32 + lane; each slot = 16 bf16 = 32 bytes
  for (int s = tid; s < 32 * 2 * 32; s += 128) {
    const int kt  = s >> 6;
    const int rem = s & 63;
    const int j   = rem >> 5;
    const int ln  = rem & 31;
    const int n   = n0 + j * 16 + (ln & 15);
    const int k   = kt * 32 + (ln >> 4) * 16;
    const __bf16* src = WhT + (size_t)n * HID + k;
    __bf16* dst = ldsB + (size_t)s * 16;
    *reinterpret_cast<v8bf*>(dst)     = *reinterpret_cast<const v8bf*>(src);
    *reinterpret_cast<v8bf*>(dst + 8) = *reinterpret_cast<const v8bf*>(src + 8);
  }
  __syncthreads();

  for (int t = 0; t < SEQ; ++t) {
    v8f acc[2];
#pragma unroll
    for (int j = 0; j < 2; ++j) acc[j] = zero_acc();

    const __bf16* Hp = H + (size_t)t * STEPEL;
    for (int k0 = 0; k0 < HID; k0 += 32) {
      v16bf af = load_A_frag(Hp, m0 + lm, HID, k0, half);
      const int kt = k0 >> 5;
#pragma unroll
      for (int j = 0; j < 2; ++j) {
        const __bf16* p = ldsB + ((size_t)(kt * 2 + j) * 32 + lane) * 16;
        acc[j] = wmma_bf16(af, load_frag16(p, p + 8), acc[j]);
      }
    }

    __bf16* Hc = H + (size_t)(t + 1) * STEPEL;
    const float* Ap = Apre + (size_t)t * STEPEL;
#pragma unroll
    for (int j = 0; j < 2; ++j) {
      const int n = n0 + j * 16 + lm;
#pragma unroll
      for (int r = 0; r < 8; ++r) {
        const int row = m0 + half * 8 + r;
        const float v = Ap[(size_t)row * HID + n] + acc[j][r];
        Hc[(size_t)row * HID + n] = (__bf16)tanhf(v);
      }
    }

    // grid-wide barrier for step t: release stores, arrive, spin-acquire
    __syncthreads();
    if (tid == 0) {
      __threadfence();
      __hip_atomic_fetch_add(&ctr[t], 1, __ATOMIC_RELEASE, __HIP_MEMORY_SCOPE_AGENT);
      while (__hip_atomic_load(&ctr[t], __ATOMIC_ACQUIRE,
                               __HIP_MEMORY_SCOPE_AGENT) < nwg) { }
    }
    __syncthreads();
  }
}

// ---------------- small utility kernels ----------------
__global__ void zero_ctr(int* __restrict__ c, int n) {
  const int i = blockIdx.x * blockDim.x + threadIdx.x;
  if (i < n) c[i] = 0;
}

__global__ void cvt_bf16(const float* __restrict__ s, __bf16* __restrict__ d, long n) {
  long i = (long)blockIdx.x * blockDim.x + threadIdx.x;
  const long stride = (long)gridDim.x * blockDim.x;
  for (; i < n; i += stride) d[i] = (__bf16)s[i];
}

// W [L][K][N] (f32) -> WT [L][N][K] (bf16), K = N = 1024
__global__ void transpose_cvt(const float* __restrict__ W, __bf16* __restrict__ WT,
                              long total) {
  long i = (long)blockIdx.x * blockDim.x + threadIdx.x;
  const long stride = (long)gridDim.x * blockDim.x;
  for (; i < total; i += stride) {
    const long k = i & 1023;
    const long n = (i >> 10) & 1023;
    const long j = i >> 20;
    WT[i] = (__bf16)W[(j << 20) + (k << 10) + n];
  }
}

__global__ void gather_emb(const int* __restrict__ tok, const float* __restrict__ emb,
                           __bf16* __restrict__ X) {
  const int row = blockIdx.x;                  // t*64 + b, 0..8191
  const float* e = emb + (size_t)tok[row] * HID;
  __bf16* x = X + (size_t)row * HID;
  for (int k = threadIdx.x; k < HID; k += blockDim.x) x[k] = (__bf16)e[k];
}

__global__ void init_h(const float* __restrict__ hid, __bf16* __restrict__ H1,
                       __bf16* __restrict__ H2) {
  const int i = blockIdx.x * blockDim.x + threadIdx.x;   // < 131072
  if (i < STEPEL) H1[i] = (__bf16)hid[i];
  else            H2[i - STEPEL] = (__bf16)hid[i];
}

__global__ void final_h(const __bf16* __restrict__ H1, const __bf16* __restrict__ H2,
                        float* __restrict__ out) {
  const int i = blockIdx.x * blockDim.x + threadIdx.x;   // < 131072
  if (i < STEPEL) out[i] = (float)H1[i];
  else            out[i] = (float)H2[i - STEPEL];
}

// ---------------- launch ----------------
extern "C" void kernel_launch(void* const* d_in, const int* in_sizes, int n_in,
                              void* d_out, int out_size, void* d_ws, size_t ws_size,
                              hipStream_t stream) {
  const int*   tok  = (const int*)  d_in[0];   // [128,64] int32
  const float* hid0 = (const float*)d_in[1];   // [2,64,1024]
  const float* emb  = (const float*)d_in[2];   // [10000,1024]
  const float* Wx   = (const float*)d_in[3];   // [2,1024,1024]
  const float* Wh   = (const float*)d_in[4];   // [2,1024,1024]
  const float* bia  = (const float*)d_in[5];   // [2,1024]
  const float* fcW  = (const float*)d_in[6];   // [10000,1024]
  const float* fcb  = (const float*)d_in[7];   // [10000]
  float* out = (float*)d_out;                  // logits [8192,10000] ++ h_final [2,64,1024]

  // workspace carve-out (~147 MB total)
  char* ws = (char*)d_ws;
  size_t off = 0;
  int* ctr = (int*)(ws + off);                       off += 4096;
  __bf16* WxT  = (__bf16*)(ws + off);                off += 2ull * HID * HID * 2;
  __bf16* WhT  = (__bf16*)(ws + off);                off += 2ull * HID * HID * 2;
  __bf16* fcWT = (__bf16*)(ws + off);                off += (size_t)VOCAB * HID * 2;
  __bf16* X    = (__bf16*)(ws + off);                off += (size_t)ROWS * HID * 2;
  __bf16* H1   = (__bf16*)(ws + off);                off += (size_t)(SEQ + 1) * STEPEL * 2;
  __bf16* H2   = (__bf16*)(ws + off);                off += (size_t)(SEQ + 1) * STEPEL * 2;
  float*  A1   = (float*)(ws + off);                 off += (size_t)ROWS * HID * 4;
  float*  A2   = (float*)(ws + off);                 off += (size_t)ROWS * HID * 4;

  // prep: counters, weight conversion (transposed to [N][K]), gather, h0
  zero_ctr<<<1, 256, 0, stream>>>(ctr, 2 * SEQ);
  transpose_cvt<<<2048, 256, 0, stream>>>(Wx, WxT, 2ll * HID * HID);
  transpose_cvt<<<2048, 256, 0, stream>>>(Wh, WhT, 2ll * HID * HID);
  cvt_bf16<<<4096, 256, 0, stream>>>(fcW, fcWT, (long)VOCAB * HID);
  gather_emb<<<ROWS, 256, 0, stream>>>(tok, emb, X);
  init_h<<<512, 256, 0, stream>>>(hid0, H1, H2);

  const dim3 blk(256);
  // layer 0 input contribution for all timesteps: A1 = X @ Wx0^T(t) + b0
  gemm_bf16_f32<<<dim3(HID / 128, ROWS / 128), blk, 0, stream>>>(
      X, WxT, bia, A1, HID, HID, HID);
  // layer 0 recurrence (persistent, grid-barriered, Wh in LDS)
  rnn_recur<<<32, 128, 0, stream>>>(WhT, A1, H1, ctr, 32);
  // layer 1 input contribution: A2 = H1 @ Wx1^T(t) + b1
  gemm_bf16_f32<<<dim3(HID / 128, ROWS / 128), blk, 0, stream>>>(
      H1 + STEPEL, WxT + (size_t)HID * HID, bia + HID, A2, HID, HID, HID);
  // layer 1 recurrence
  rnn_recur<<<32, 128, 0, stream>>>(WhT + (size_t)HID * HID, A2, H2, ctr + SEQ, 32);
  // vocab projection: logits = H2 @ fcW^T + fcb  (N = 10000, edge-guarded)
  gemm_bf16_f32<<<dim3((VOCAB + 127) / 128, ROWS / 128), blk, 0, stream>>>(
      H2 + STEPEL, fcWT, fcb, out, VOCAB, HID, VOCAB);
  // h_final (f32) appended after logits
  final_h<<<512, 256, 0, stream>>>(H1 + (size_t)SEQ * STEPEL,
                                   H2 + (size_t)SEQ * STEPEL,
                                   out + (size_t)ROWS * VOCAB);
  (void)in_sizes; (void)n_in; (void)out_size; (void)ws_size;
}
